// FastQuantumAttention_38826504356525
// MI455X (gfx1250) — matrix operand order repaired and checked by reading, not compiled, not run on hardware
//
#include <hip/hip_runtime.h>

typedef _Float16 v16h __attribute__((ext_vector_type(16)));
typedef _Float16 v8h  __attribute__((ext_vector_type(8)));
typedef _Float16 v4h  __attribute__((ext_vector_type(4)));
typedef float    v8f  __attribute__((ext_vector_type(8)));

#define XSTR 136   // padded half stride for Xh rows (128 + 8) -> 272B = 17*16B
#define ASTR 40    // padded half stride for attn rows (32 + 8) -> 80B  = 5*16B

__device__ inline v16h pack16(v8h lo, v8h hp) {
    v16h r;
#pragma unroll
    for (int k = 0; k < 8; ++k) { r[k] = lo[k]; r[k + 8] = hp[k]; }
    return r;
}

__global__ __launch_bounds__(128) void fqa_kernel(const float* __restrict__ xg,
                                                  const float* __restrict__ pg,
                                                  float* __restrict__ og)
{
    __shared__ _Float16 sXh[4][32 * XSTR];   // per-wave Xh (32 heads x 128, f16)
    __shared__ _Float16 sAt[4][32 * ASTR];   // per-wave attn (32 x 32, f16)

    const int lane = threadIdx.x & 31;
    const int wid  = threadIdx.x >> 5;
    const int lh   = lane & 15;   // lane within 16-group (= N index / M row select)
    const int hi   = lane >> 4;   // which 16-lane half
    const int token = blockIdx.x * 4 + wid;   // grid sized exactly: B*L/4 blocks

    _Float16* Xh = sXh[wid];
    _Float16* At = sAt[wid];

    const float4* x4 = (const float4*)xg + (size_t)token * 1024;  // 4096 f32 / token
    const float4* p4 = (const float4*)pg;                          // 32x128 f32

    // ---- Phase 1: stream token row, modulate by patterns, stage f16 Xh in LDS ----
    // lane-major float4 -> each global_load_b128 instruction covers 512 contiguous B.
    // iteration j == head j; lane l holds d = 4l..4l+3.
#pragma unroll
    for (int j = 0; j < 32; ++j) {
        float4 xv = x4[j * 32 + lane];
        float4 pv = p4[j * 32 + lane];
        v4h h;
        h[0] = (_Float16)(xv.x * pv.x);
        h[1] = (_Float16)(xv.y * pv.y);
        h[2] = (_Float16)(xv.z * pv.z);
        h[3] = (_Float16)(xv.w * pv.w);
        *(v4h*)(&Xh[j * XSTR + lane * 4]) = h;   // 8B store, conflict-free
    }
    __syncthreads();

    // ---- Phase 2: S = Xh * Xh^T (32x32) via 2x2 tiles, K=128 in 4 steps ----
    v8f acc[2][2];
#pragma unroll
    for (int i = 0; i < 2; ++i)
#pragma unroll
        for (int j = 0; j < 2; ++j) acc[i][j] = v8f{};

#pragma unroll
    for (int k0 = 0; k0 < 128; k0 += 32) {
        v16h a[2], b[2];
        // A fragment (16x32 f16): lanes<16: K = k0+0..15 ; lanes>=16: K = k0+16..31
#pragma unroll
        for (int i = 0; i < 2; ++i) {
            const _Float16* r = &Xh[(i * 16 + lh) * XSTR + k0 + hi * 16];
            a[i] = pack16(*(const v8h*)r, *(const v8h*)(r + 8));
        }
        // B fragment (32x16 f16): lane n holds column n = Xh row (j*16+n), K contiguous
#pragma unroll
        for (int j = 0; j < 2; ++j) {
            const _Float16* r = &Xh[(j * 16 + lh) * XSTR + k0 + hi * 16];
            b[j] = pack16(*(const v8h*)r, *(const v8h*)(r + 8));
        }
#pragma unroll
        for (int i = 0; i < 2; ++i)
#pragma unroll
            for (int j = 0; j < 2; ++j)
                acc[i][j] = __builtin_amdgcn_wmma_f32_16x16x32_f16(
                    false, a[i], false, b[j], (short)0, acc[i][j], false, false);
    }

    // ---- Phase 3: row softmax over g (N axis) in the WMMA C layout ----
    // VGPR v: lanes<16 hold row M=v (N=lane), lanes>=16 hold row M=v+8.
    const float scale = 0.08838834764831845f;   // 1/sqrt(128)
#pragma unroll
    for (int i = 0; i < 2; ++i) {
#pragma unroll
        for (int v = 0; v < 8; ++v) {
            float s0 = acc[i][0][v] * scale;
            float s1 = acc[i][1][v] * scale;
            float m = fmaxf(s0, s1);
#pragma unroll
            for (int msk = 1; msk < 16; msk <<= 1)
                m = fmaxf(m, __shfl_xor(m, msk, 32));
            float e0 = __expf(s0 - m);
            float e1 = __expf(s1 - m);
            float t = e0 + e1;
#pragma unroll
            for (int msk = 1; msk < 16; msk <<= 1)
                t += __shfl_xor(t, msk, 32);
            float r = 1.0f / t;
            int row = i * 16 + v + hi * 8;
            At[row * ASTR + lh]      = (_Float16)(e0 * r);
            At[row * ASTR + 16 + lh] = (_Float16)(e1 * r);
        }
    }
    __syncthreads();

    // ---- Phase 4: out = A(32x32) * Xh(32x128), 2x8 tiles, K=32 ----
    v16h a2[2];
#pragma unroll
    for (int i = 0; i < 2; ++i) {
        const _Float16* r = &At[(i * 16 + lh) * ASTR + hi * 16];
        a2[i] = pack16(*(const v8h*)r, *(const v8h*)(r + 8));
    }

    const size_t obase = (size_t)token * 4096;
#pragma unroll
    for (int n0 = 0; n0 < 8; ++n0) {
        // B fragment: column n = Xh[:, n0*16+n] over heads (strided gather;
        // lane pairs read the same LDS dword -> broadcast, conflict-free)
        v16h b2;
#pragma unroll
        for (int k = 0; k < 16; ++k)
            b2[k] = Xh[(hi * 16 + k) * XSTR + n0 * 16 + lh];
#pragma unroll
        for (int i = 0; i < 2; ++i) {
            v8f z = v8f{};
            v8f o = __builtin_amdgcn_wmma_f32_16x16x32_f16(
                false, a2[i], false, b2, (short)0, z, false, false);
#pragma unroll
            for (int v = 0; v < 8; ++v)
                og[obase + (size_t)((i * 16 + v + hi * 8) * 128 + n0 * 16 + lh)] = o[v];
        }
    }
}

extern "C" void kernel_launch(void* const* d_in, const int* in_sizes, int n_in,
                              void* d_out, int out_size, void* d_ws, size_t ws_size,
                              hipStream_t stream) {
    const float* x = (const float*)d_in[0];        // [4, 4096, 4096] f32
    const float* p = (const float*)d_in[1];        // [32, 128] f32
    float* out = (float*)d_out;                    // [4, 4096, 4096] f32

    const int tokens = in_sizes[0] / 4096;         // B*L = 16384
    dim3 grid(tokens / 4);                         // 4 waves (tokens) per block
    dim3 block(128);
    fqa_kernel<<<grid, block, 0, stream>>>(x, p, out);
}
